// TA5Block_82506321756931
// MI455X (gfx1250) — matrix operand order repaired and checked by reading, not compiled, not run on hardware
//
#include <hip/hip_runtime.h>

// ---------------------------------------------------------------------------
// TA5Block fused pipeline for MI455X (gfx1250, wave32, WMMA)
// Shapes: B=8, TP=3, TF=4, C=256, H=W=64  -> N = 131072 pixels for the 1x1 conv
// ---------------------------------------------------------------------------

typedef __attribute__((ext_vector_type(16))) __bf16 v16bf;
typedef __attribute__((ext_vector_type(8)))  __bf16 v8bf;
typedef __attribute__((ext_vector_type(8)))  float  v8f;

#define C_DIM 256
#define HW    4096
#define BT    32            // B*(TP+1)
#define BN_EPS 0.001f

// fp32 -> bf16 (round to nearest even), bit-twiddled (no __bf16 scalar ops)
__device__ __forceinline__ unsigned short f2bf(float x) {
    unsigned int u = __float_as_uint(x);
    u += 0x7FFFu + ((u >> 16) & 1u);
    return (unsigned short)(u >> 16);
}
__device__ __forceinline__ float bf2f(unsigned short h) {
    return __uint_as_float(((unsigned int)h) << 16);
}
__device__ __forceinline__ float silu(float x) {
    return x / (1.0f + __expf(-x));
}

// ---------------------------------------------------------------------------
// Kernel 0: fold BN scale into conv weights, split hi/lo bf16, and swizzle into
// WMMA A-fragment order:  Whi[((mt*8+ks)*32 + lane)*16 + e]
// A-layout (16-bit A 16x32, ISA 7.12.2): lane<16 -> M=lane, K groups {0..7,16..23};
// lane>=16 -> M=lane-16, K groups {8..15,24..31}; VGPR pairs pack (2k,2k+1).
// ---------------------------------------------------------------------------
__global__ __launch_bounds__(256) void prep_weights(
    const float* __restrict__ conv_w, const float* __restrict__ g,
    const float* __restrict__ be, const float* __restrict__ mn,
    const float* __restrict__ vr,
    unsigned short* __restrict__ Whi, unsigned short* __restrict__ Wlo,
    float* __restrict__ shiftp)
{
    int gid = blockIdx.x * 256 + threadIdx.x;     // 0 .. 65535
    int e  = gid & 15;
    int L  = (gid >> 4) & 31;
    int ks = (gid >> 9) & 7;
    int mt = gid >> 12;
    int m    = L & 15;
    int koff = (L >> 4) * 8;
    int v    = e >> 1, half = e & 1;
    int k    = (v < 4 ? 2 * v : 16 + 2 * (v - 4)) + koff + half;
    int o = mt * 16 + m;
    int c = ks * 32 + k;
    float scale = g[o] * rsqrtf(vr[o] + BN_EPS);
    float w = conv_w[o * C_DIM + c] * scale;
    unsigned short h = f2bf(w);
    unsigned short l = f2bf(w - bf2f(h));
    Whi[gid] = h;
    Wlo[gid] = l;
    if (blockIdx.x == 0) {
        int o2 = threadIdx.x;
        float s2 = g[o2] * rsqrtf(vr[o2] + BN_EPS);
        shiftp[o2] = be[o2] - mn[o2] * s2;
    }
}

// ---------------------------------------------------------------------------
// Kernel 1: 1x1 conv GEMM (256x256 x pixels) + BN shift + SiLU via bf16 hi/lo
// split WMMA.  Grid: 2048 blocks = 32 (b,t) x 64 pixel-tiles of 64 pixels.
// Block: 256 threads = 8 waves; wave (mhalf, ngroup) owns 8 M-tiles x 16 pixels.
// ---------------------------------------------------------------------------
__global__ __launch_bounds__(256) void conv_bn_silu_wmma(
    const float* __restrict__ feature,
    const unsigned short* __restrict__ Whi, const unsigned short* __restrict__ Wlo,
    const float* __restrict__ shiftp, float* __restrict__ featp)
{
    __shared__ unsigned short FThi[64 * 40];   // [pixel][c] bf16 hi, row stride 40 (pad)
    __shared__ unsigned short FTlo[64 * 40];   // [pixel][c] bf16 lo

    const int tid    = threadIdx.x;
    const int lane   = tid & 31;
    const int wave   = tid >> 5;
    const int ngroup = wave & 3;     // which 16-pixel column group
    const int mhalf  = wave >> 2;    // which 8 M-tiles
    const int bt     = blockIdx.x >> 6;
    const int ptile  = blockIdx.x & 63;
    const int pblock = ptile * 64;

    const float* fbase = feature + (size_t)bt * C_DIM * HW + pblock;

    // Accumulators seeded with the BN shift (bn_beta - bn_mean*scale).
    const int ohalf = (lane >> 4) << 3;       // C/D rows: VGPR j -> M = j (+8 for hi lanes)
    v8f acc[8];
#pragma unroll
    for (int mi = 0; mi < 8; ++mi) {
        const int obase = (mhalf * 8 + mi) * 16 + ohalf;
#pragma unroll
        for (int j = 0; j < 8; ++j) acc[mi][j] = shiftp[obase + j];
    }

    // Prefetch k-chunk 0: 32 channels x 64 pixels (fp32), 2 float4 per thread.
    const int r0 = tid >> 4;             // 0..15
    const int p4 = (tid & 15) * 4;       // 0..60
    float4 rg[2];
    rg[0] = *(const float4*)(fbase + (size_t)(r0) * HW + p4);
    rg[1] = *(const float4*)(fbase + (size_t)(r0 + 16) * HW + p4);

    // B fragment source (16-bit B 32x16): lane<16 -> col=lane, K={0..7,16..23};
    // lane>=16 -> col=lane-16, K={8..15,24..31}; two runs of 8 bf16 per lane.
    const int pw   = ngroup * 16 + (lane & 15);
    const int coff = (lane >> 4) << 3;   // 0 or 8
    const unsigned short* bph = &FThi[pw * 40 + coff];
    const unsigned short* bpl = &FTlo[pw * 40 + coff];

    for (int ks = 0; ks < 8; ++ks) {
        __syncthreads();
        // Store prefetched chunk to LDS, transposed, as bf16 hi/lo pair.
#pragma unroll
        for (int i = 0; i < 2; ++i) {
            const int cl = r0 + 16 * i;
            float vv[4] = {rg[i].x, rg[i].y, rg[i].z, rg[i].w};
#pragma unroll
            for (int j = 0; j < 4; ++j) {
                unsigned short h = f2bf(vv[j]);
                FThi[(p4 + j) * 40 + cl] = h;
                FTlo[(p4 + j) * 40 + cl] = f2bf(vv[j] - bf2f(h));
            }
        }
        __syncthreads();
        if (ks < 7) {   // prefetch next chunk; waits deferred to next LDS-store
            rg[0] = *(const float4*)(fbase + (size_t)((ks + 1) * 32 + r0) * HW + p4);
            rg[1] = *(const float4*)(fbase + (size_t)((ks + 1) * 32 + r0 + 16) * HW + p4);
        }

        v8bf bh0 = *(const v8bf*)(bph);
        v8bf bh1 = *(const v8bf*)(bph + 16);
        v8bf bl0 = *(const v8bf*)(bpl);
        v8bf bl1 = *(const v8bf*)(bpl + 16);
        v16bf bhi = __builtin_shufflevector(bh0, bh1, 0,1,2,3,4,5,6,7,8,9,10,11,12,13,14,15);
        v16bf blo = __builtin_shufflevector(bl0, bl1, 0,1,2,3,4,5,6,7,8,9,10,11,12,13,14,15);

#pragma unroll
        for (int mi = 0; mi < 8; ++mi) {
            const int mt = mhalf * 8 + mi;
            const size_t off = ((size_t)(mt * 8 + ks) * 32 + lane) * 16;
            v16bf ahi = *(const v16bf*)(Whi + off);
            v16bf alo = *(const v16bf*)(Wlo + off);
            // W*F ~= Wh*Fh + Wl*Fh + Wh*Fl  (lo*lo dropped), fp32 accumulate
            acc[mi] = __builtin_amdgcn_wmma_f32_16x16x32_bf16(false, ahi, false, bhi,
                                                              (short)0, acc[mi], false, false);
            acc[mi] = __builtin_amdgcn_wmma_f32_16x16x32_bf16(false, alo, false, bhi,
                                                              (short)0, acc[mi], false, false);
            acc[mi] = __builtin_amdgcn_wmma_f32_16x16x32_bf16(false, ahi, false, blo,
                                                              (short)0, acc[mi], false, false);
        }
    }

    // Epilogue: SiLU and store feature_p.
    const int pixg = pblock + ngroup * 16 + (lane & 15);
    float* outb = featp + (size_t)bt * C_DIM * HW + pixg;
#pragma unroll
    for (int mi = 0; mi < 8; ++mi) {
        const int obase = (mhalf * 8 + mi) * 16 + ohalf;
#pragma unroll
        for (int j = 0; j < 8; ++j) {
            float x = acc[mi][j];
            outb[(size_t)(obase + j) * HW] = silu(x);
        }
    }
}

// ---------------------------------------------------------------------------
// Kernel 2: spatial max+mean pools. t<3: over feature_p (pool_p);
// t==3: over the RAW last feature frame (pool_f).  Block per (b,t,c).
// pools layout: [(b*4+t)*512 + c] = max, [+256+c] = mean.
// ---------------------------------------------------------------------------
__global__ __launch_bounds__(256) void pool_kernel(
    const float* __restrict__ feature, const float* __restrict__ featp,
    float* __restrict__ pools)
{
    __shared__ float smax[256];
    __shared__ float ssum[256];
    const int c = blockIdx.x & 255;
    const int t = (blockIdx.x >> 8) & 3;
    const int b = blockIdx.x >> 10;
    const float* src = (t < 3 ? featp : feature) + ((size_t)(b * 4 + t) * C_DIM + c) * HW;
    float mx = -3.402823466e38f, sm = 0.0f;
    for (int i = threadIdx.x; i < 1024; i += 256) {
        float4 v = *(const float4*)(src + i * 4);
        mx = fmaxf(mx, fmaxf(fmaxf(v.x, v.y), fmaxf(v.z, v.w)));
        sm += (v.x + v.y) + (v.z + v.w);
    }
    smax[threadIdx.x] = mx;
    ssum[threadIdx.x] = sm;
    __syncthreads();
    for (int s = 128; s > 0; s >>= 1) {
        if (threadIdx.x < s) {
            smax[threadIdx.x] = fmaxf(smax[threadIdx.x], smax[threadIdx.x + s]);
            ssum[threadIdx.x] += ssum[threadIdx.x + s];
        }
        __syncthreads();
    }
    if (threadIdx.x == 0) {
        pools[(size_t)(b * 4 + t) * 512 + c]       = smax[0];
        pools[(size_t)(b * 4 + t) * 512 + 256 + c] = ssum[0] * (1.0f / 4096.0f);
    }
}

// ---------------------------------------------------------------------------
// Kernel 3: all the tiny vector math -> attn_w[b][f][p] (96 floats).
// One block per batch.
// ---------------------------------------------------------------------------
__global__ __launch_bounds__(256) void attn_small(
    const float* __restrict__ pools, const float* __restrict__ past_tc,
    const float* __restrict__ fut_tc, const float* __restrict__ w_tc,
    const float* __restrict__ b_tc, const float* __restrict__ w_in,
    const float* __restrict__ b_in, const float* __restrict__ w_q,
    const float* __restrict__ b_q, const float* __restrict__ w_k,
    const float* __restrict__ b_k, float* __restrict__ attnw)
{
    __shared__ float pl[4 * 512];    // pools for this b
    __shared__ float ai[7 * 256];    // [0..2]=attn_in_p(t), [3..6]=attn_in_f(f)
    __shared__ float qk[7 * 256];    // [0..3]=q(f), [4..6]=k(t)
    const int b = blockIdx.x, tid = threadIdx.x;
    for (int i = tid; i < 2048; i += 256) pl[i] = pools[(size_t)b * 2048 + i];
    __syncthreads();

    const int c = tid;
    const float wtc = w_tc[c], btc = b_tc[c], bin = b_in[c];
    for (int t = 0; t < 3; ++t) {
        float s = bin;
        for (int j = 0; j < 512; ++j) s += w_in[c * 512 + j] * pl[t * 512 + j];
        ai[t * 256 + c] = silu(s) + tanhf(past_tc[b * 3 + t] * wtc + btc);
    }
    {
        float s = bin;
        for (int j = 0; j < 512; ++j) s += w_in[c * 512 + j] * pl[3 * 512 + j];
        float sl = silu(s);
        for (int f = 0; f < 4; ++f)
            ai[(3 + f) * 256 + c] = sl + tanhf(fut_tc[b * 4 + f] * wtc + btc);
    }
    __syncthreads();

    for (int f = 0; f < 4; ++f) {
        float s = b_q[c];
        for (int d = 0; d < 256; ++d) s += ai[(3 + f) * 256 + d] * w_q[c * 256 + d];
        qk[f * 256 + c] = s;
    }
    for (int t = 0; t < 3; ++t) {
        float s = b_k[c];
        for (int d = 0; d < 256; ++d) s += ai[t * 256 + d] * w_k[c * 256 + d];
        qk[(4 + t) * 256 + c] = s;
    }
    __syncthreads();

    if (tid < 12) {
        const int f = tid / 3, p = tid % 3;
        float s = 0.0f;
        for (int d = 0; d < 256; ++d) s += qk[f * 256 + d] * qk[(4 + p) * 256 + d];
        attnw[(b * 4 + f) * 3 + p] = s * 0.5f;   // 1/sqrt(TF=4)
    }
}

// ---------------------------------------------------------------------------
// Kernel 4: streaming epilogue.
// out[b,f,c,hw] = last + (sum_p aw[f][p]*(fp_last - fp_past[p])) * p_attn[c]/3
// ---------------------------------------------------------------------------
__global__ __launch_bounds__(256) void apply_attn(
    const float* __restrict__ feature, const float* __restrict__ featp,
    const float* __restrict__ attnw, const float* __restrict__ p_attn,
    float* __restrict__ out)
{
    const int ch = blockIdx.x & 3;
    const int c  = (blockIdx.x >> 2) & 255;
    const int b  = blockIdx.x >> 10;
    const int hw = ch * 1024 + threadIdx.x * 4;
    const size_t hwoff = (size_t)c * HW + hw;

    float aw[4][3];
#pragma unroll
    for (int f = 0; f < 4; ++f)
#pragma unroll
        for (int p = 0; p < 3; ++p) aw[f][p] = attnw[(b * 4 + f) * 3 + p];

    float4 last4 = *(const float4*)(feature + (size_t)(b * 4 + 3) * C_DIM * HW + hwoff);
    float4 fp3   = *(const float4*)(featp   + (size_t)(b * 4 + 3) * C_DIM * HW + hwoff);
    float4 fpp[3];
#pragma unroll
    for (int p = 0; p < 3; ++p)
        fpp[p] = *(const float4*)(featp + (size_t)(b * 4 + p) * C_DIM * HW + hwoff);

    const float pa = p_attn[c] * (1.0f / 3.0f);
    const float* lastA = (const float*)&last4;
    const float* fp3A  = (const float*)&fp3;
#pragma unroll
    for (int f = 0; f < 4; ++f) {
        float4 o4;
        float* oA = (float*)&o4;
#pragma unroll
        for (int e = 0; e < 4; ++e) {
            float a = 0.0f;
#pragma unroll
            for (int p = 0; p < 3; ++p)
                a += aw[f][p] * (fp3A[e] - ((const float*)&fpp[p])[e]);
            oA[e] = lastA[e] + a * pa;
        }
        *(float4*)(out + (size_t)(b * 4 + f) * C_DIM * HW + hwoff) = o4;
    }
}

// ---------------------------------------------------------------------------
extern "C" void kernel_launch(void* const* d_in, const int* in_sizes, int n_in,
                              void* d_out, int out_size, void* d_ws, size_t ws_size,
                              hipStream_t stream) {
    (void)in_sizes; (void)n_in; (void)out_size; (void)ws_size;
    const float* feature  = (const float*)d_in[0];
    const float* past_tc  = (const float*)d_in[1];
    const float* fut_tc   = (const float*)d_in[2];
    const float* w_tc     = (const float*)d_in[3];
    const float* b_tc     = (const float*)d_in[4];
    const float* w_in     = (const float*)d_in[5];
    const float* b_in     = (const float*)d_in[6];
    const float* conv_w   = (const float*)d_in[7];
    const float* bn_gamma = (const float*)d_in[8];
    const float* bn_beta  = (const float*)d_in[9];
    const float* bn_mean  = (const float*)d_in[10];
    const float* bn_var   = (const float*)d_in[11];
    const float* w_q      = (const float*)d_in[12];
    const float* b_q      = (const float*)d_in[13];
    const float* w_k      = (const float*)d_in[14];
    const float* b_k      = (const float*)d_in[15];
    const float* p_attn   = (const float*)d_in[16];
    float* out = (float*)d_out;

    // Workspace layout
    const size_t FEATP_BYTES = (size_t)BT * C_DIM * HW * sizeof(float);   // 134,217,728
    char* ws = (char*)d_ws;
    float*          featp  = (float*)ws;
    unsigned short* Whi    = (unsigned short*)(ws + FEATP_BYTES);         // 65536 bf16
    unsigned short* Wlo    = Whi + 65536;
    float*          shiftp = (float*)(ws + FEATP_BYTES + 262144);         // 256 f32
    float*          pools  = shiftp + 256;                                // 16384 f32
    float*          attnw  = pools + 16384;                               // 96 f32

    prep_weights<<<256, 256, 0, stream>>>(conv_w, bn_gamma, bn_beta, bn_mean, bn_var,
                                          Whi, Wlo, shiftp);
    conv_bn_silu_wmma<<<2048, 256, 0, stream>>>(feature, Whi, Wlo, shiftp, featp);
    pool_kernel<<<8192, 256, 0, stream>>>(feature, featp, pools);
    attn_small<<<8, 256, 0, stream>>>(pools, past_tc, fut_tc, w_tc, b_tc, w_in, b_in,
                                      w_q, b_q, w_k, b_k, attnw);
    apply_attn<<<8192, 256, 0, stream>>>(feature, featp, attnw, p_attn, out);
}
